// NWJ_22832046146016
// MI455X (gfx1250) — matrix operand order repaired and checked by reading, not compiled, not run on hardware
//
#include <hip/hip_runtime.h>
#include <hip/hip_bf16.h>

typedef __attribute__((ext_vector_type(16))) _Float16 v16h;
typedef __attribute__((ext_vector_type(8)))  _Float16 v8h;
typedef __attribute__((ext_vector_type(8)))  float    v8f;

#define NS   1024
#define XDIM 128
#define HID  256

// Concatenate two v8h runs into a v16h WMMA A/B fragment (register pairing only).
__device__ __forceinline__ v16h cat16(v8h a, v8h b) {
    return __builtin_shufflevector(a, b, 0, 1, 2, 3, 4, 5, 6, 7,
                                         8, 9, 10, 11, 12, 13, 14, 15);
}

// relu(p + q) in packed f16 (v_pk_add_f16 + v_pk_max_num_f16), concat to v16h.
__device__ __forceinline__ v16h relu_cat(v8h p0, v8h p1, v8h q0, v8h q1) {
    const v8h z = {};
    return cat16(__builtin_elementwise_max(p0 + q0, z),
                 __builtin_elementwise_max(p1 + q1, z));
}

// Per-lane async stage of a 16x256 f16 tile into padded LDS (row stride 264
// halves). 2 x global_load_async_to_lds_b128 per lane, tracked by ASYNCcnt.
__device__ __forceinline__ void stage_tile_async(const _Float16* __restrict__ gsrc,
                                                 _Float16* ldst, int t) {
#pragma unroll
    for (int q = 0; q < 2; ++q) {
        const int f   = t + q * 256;          // 16B-chunk index, 0..511
        const int row = f >> 5;               // 32 chunks per 256-half row
        const int c   = f & 31;
        const unsigned lds_off =
            (unsigned)(uintptr_t)(&ldst[row * 264 + c * 8]);    // low 32b of flat = LDS offset
        const unsigned long long ga =
            (unsigned long long)(uintptr_t)(gsrc + (size_t)f * 8);
        asm volatile("global_load_async_to_lds_b128 %0, %1, off"
                     :: "v"(lds_off), "v"(ga) : "memory");
    }
}

// ---------------------------------------------------------------------------
// Kernel 0a: W1t[h][k] = (f16) W1[k][h]  (256x256) — B-operand panel for k1.
// ---------------------------------------------------------------------------
__global__ __launch_bounds__(256) void k0_w1t(const float* __restrict__ W1,
                                              _Float16* __restrict__ W1t) {
    const int h = blockIdx.x;
    const int t = threadIdx.x;
    W1t[(size_t)h * HID + t] = (_Float16)W1[(size_t)t * HID + h];
}

// ---------------------------------------------------------------------------
// Kernel 0b: one-shot f32 -> f16 conversion of x and y (A operands for k1).
// 128 blocks x 256 threads, one float4 of each per thread.
// ---------------------------------------------------------------------------
__global__ __launch_bounds__(256) void k0_cvt(const float* __restrict__ x,
                                              const float* __restrict__ y,
                                              _Float16* __restrict__ xh,
                                              _Float16* __restrict__ yh) {
    const int f = blockIdx.x * 256 + threadIdx.x;     // float4 index, 0..32767
    float4 a = ((const float4*)x)[f];
    float4 b = ((const float4*)y)[f];
    v8h o;
    o[0] = (_Float16)a.x; o[1] = (_Float16)a.y; o[2] = (_Float16)a.z; o[3] = (_Float16)a.w;
    o[4] = (_Float16)b.x; o[5] = (_Float16)b.y; o[6] = (_Float16)b.z; o[7] = (_Float16)b.w;
    *(uint2*)(xh + (size_t)f * 4) = ((uint2*)&o)[0];
    *(uint2*)(yh + (size_t)f * 4) = ((uint2*)&o)[1];
}

// ---------------------------------------------------------------------------
// Kernel 1: pxbh = f16(x @ W1[:128] + b1) ; pyh = f16(y @ W1[128:])
// One wave per 16x16 output tile, K = 128 -> 4 WMMAs (f32 accumulate).
// A-frag (16-bit A 16x32): lane M=lane&15, half=lane>>4:
//   e 0..15 -> K = kb + 16*half + e (same K order as B)
// B-frag (16-bit B 32x16): lane col N=lane&15, element e -> K = kb + e + 16*half
// ---------------------------------------------------------------------------
__global__ __launch_bounds__(32) void k1_gemm(const _Float16* __restrict__ xh,
                                              const _Float16* __restrict__ yh,
                                              const _Float16* __restrict__ W1t,
                                              const float* __restrict__ b1,
                                              _Float16* __restrict__ pxbh,
                                              _Float16* __restrict__ pyh) {
    const int lane = threadIdx.x & 31;
    const int M    = lane & 15;       // A row within tile == D column within tile
    const int half = lane >> 4;
    const int nbase = blockIdx.x * 16;
    const int hbase = blockIdx.y * 16;
    const int mat   = blockIdx.z;     // 0 -> x path (adds b1), 1 -> y path
    const _Float16* Ain = mat ? yh : xh;
    const int wrow      = mat * XDIM; // W1 row offset (W1x vs W1y)

    const _Float16* arow = Ain + (size_t)(nbase + M) * XDIM;
    const _Float16* brow = W1t + (size_t)(hbase + M) * HID + wrow + 16 * half;

    v8f acc = {};
#pragma unroll
    for (int kb = 0; kb < XDIM; kb += 32) {
        const int k0 = kb + 16 * half;
        v16h af = cat16(*(const v8h*)(arow + k0), *(const v8h*)(arow + k0 + 8));
        v16h bf = *(const v16h*)(brow + kb);
        acc = __builtin_amdgcn_wmma_f32_16x16x32_f16(false, af, false, bf,
                                                     (short)0, acc, false, false);
    }

    const float bias = mat ? 0.0f : b1[hbase + M];
    _Float16* out = mat ? pyh : pxbh;
#pragma unroll
    for (int r = 0; r < 8; ++r) {
        const int row = nbase + r + 8 * half;       // D: VGPR r -> M = r + 8*half
        out[(size_t)row * HID + hbase + M] = (_Float16)(acc[r] + bias);
    }
}

// ---------------------------------------------------------------------------
// Kernel 2: pairwise T1 + per-row online logsumexp + diagonal T0 harvest.
// Block = 256 threads = 8 waves; wave w owns row i = blockIdx.x*8 + w.
// Double-buffered f16 j-tile pipeline staged LDS-direct with
// global_load_async_to_lds_b128; hot loop is packed-f16 VALU + WMMA, with
// py and W2 fragments held loop-invariant in registers.
// ---------------------------------------------------------------------------
__global__ __launch_bounds__(256) void k2_pair(const _Float16* __restrict__ pxbh,
                                               const _Float16* __restrict__ pyh,
                                               const float* __restrict__ W2,
                                               const float* __restrict__ b2,
                                               float* __restrict__ term) {
    __shared__ _Float16 tile[2][16 * 264] __attribute__((aligned(16)));  // padded rows
    __shared__ _Float16 pyl[8 * 256]      __attribute__((aligned(16)));
    __shared__ _Float16 w2h[256]          __attribute__((aligned(32)));

    const int t    = threadIdx.x;
    const int wave = t >> 5;
    const int lane = t & 31;
    const int M    = lane & 15;
    const int half = lane >> 4;
    const int ibase = blockIdx.x * 8;
    const int i     = ibase + wave;

    // Stage W2 (f16) and the 8 py rows (f16, vector copy) this block needs.
    w2h[t] = (_Float16)W2[t];
    *(v8h*)(&pyl[t * 8]) = *(const v8h*)(pyh + (size_t)ibase * 256 + t * 8);

    // Pipeline prologue: async-stage tiles 0 and 1.
    stage_tile_async(pxbh,                  tile[0], t);
    stage_tile_async(pxbh + (size_t)4096,   tile[1], t);

    __syncthreads();   // w2h / pyl visible to all waves

    // Loop-invariant fragments, held in registers across all 64 j-tiles:
    //  - bfrag: W2 replicated across all 16 B columns
    //  - pq:    this wave's py row, laid out to match the A-fragment K order
    v16h bfrag[8];
    v8h  pq[8][2];
    const _Float16* pyrow = &pyl[wave * 256];
#pragma unroll
    for (int kb8 = 0; kb8 < 8; ++kb8) {
        bfrag[kb8] = cat16(*(const v8h*)(&w2h[kb8 * 32 + 8 * half]), *(const v8h*)(&w2h[kb8 * 32 + 8 * half + 16]));
        const int k0 = kb8 * 32 + 8 * half;
        pq[kb8][0] = *(const v8h*)(pyrow + k0);
        pq[kb8][1] = *(const v8h*)(pyrow + k0 + 16);
    }

    const float b2v = b2[0];
    float m = -1e30f, s = 0.0f, t0 = 0.0f;

    for (int jt = 0; jt < 64; ++jt) {
        // Each lane issued 2 async loads per tile, in-order: <=2 outstanding
        // means tile[jt&1] is complete (next tile may still be in flight).
        if (jt < 63) asm volatile("s_wait_asynccnt 0x2" ::: "memory");
        else         asm volatile("s_wait_asynccnt 0x0" ::: "memory");
        __syncthreads();

        const _Float16* tl = tile[jt & 1];
        v8f acc = {};
#pragma unroll
        for (int kb8 = 0; kb8 < 8; ++kb8) {
            const int k0 = kb8 * 32 + 8 * half;
            const _Float16* tr = &tl[M * 264 + k0];
            v8h p0 = *(const v8h*)(tr);
            v8h p1 = *(const v8h*)(tr + 16);
            v16h af = relu_cat(p0, p1, pq[kb8][0], pq[kb8][1]);
            acc = __builtin_amdgcn_wmma_f32_16x16x32_f16(false, af, false, bfrag[kb8],
                                                         (short)0, acc, false, false);
        }

        // Online logsumexp over this wave's 8 D values (rows jbase + r + 8*half).
        const int jbase = jt * 16;
#pragma unroll
        for (int r = 0; r < 8; ++r) {
            const float v  = acc[r] + b2v - 1.0f;   // T1[i, jrow]
            const float mn = fmaxf(m, v);
            s = s * __expf(m - mn) + __expf(v - mn);
            m = mn;
            const int jrow = jbase + r + 8 * half;
            if (jrow == i) t0 = v + 1.0f;           // T0[i] = T1[i,i] + 1
        }
        __syncthreads();   // all waves done reading tile[jt&1]

        // Refill the just-freed buffer with tile jt+2.
        if (jt + 2 < 64)
            stage_tile_async(pxbh + (size_t)(jt + 2) * 4096, tile[jt & 1], t);
    }

    // Merge the two lane-halves (each half saw a disjoint set of j rows).
    const float m2 = __shfl_xor(m, 16, 32);
    const float s2 = __shfl_xor(s, 16, 32);
    const float mm = fmaxf(m, m2);
    const float ss = s * __expf(m - mm) + s2 * __expf(m2 - mm);
    const float lse = mm + __logf(ss);
    t0 += __shfl_xor(t0, 16, 32);

    if (lane == 0)
        term[i] = t0 - __expf(lse - __logf((float)NS));
}

// ---------------------------------------------------------------------------
// Kernel 3: out = mean(term)
// ---------------------------------------------------------------------------
__global__ __launch_bounds__(256) void k3_reduce(const float* __restrict__ term,
                                                 float* __restrict__ out) {
    __shared__ float red[256];
    const int t = threadIdx.x;
    red[t] = term[t] + term[t + 256] + term[t + 512] + term[t + 768];
    __syncthreads();
#pragma unroll
    for (int off = 128; off > 0; off >>= 1) {
        if (t < off) red[t] += red[t + off];
        __syncthreads();
    }
    if (t == 0) out[0] = red[0] * (1.0f / (float)NS);
}

extern "C" void kernel_launch(void* const* d_in, const int* in_sizes, int n_in,
                              void* d_out, int out_size, void* d_ws, size_t ws_size,
                              hipStream_t stream) {
    (void)in_sizes; (void)n_in; (void)out_size; (void)ws_size;
    const float* x  = (const float*)d_in[0];
    const float* y  = (const float*)d_in[1];
    const float* W1 = (const float*)d_in[2];
    const float* b1 = (const float*)d_in[3];
    const float* W2 = (const float*)d_in[4];
    const float* b2 = (const float*)d_in[5];

    float*    term = (float*)d_ws;                        // 1024 f32
    _Float16* pxbh = (_Float16*)(term + NS);              // 1024*256 f16 (pre_x + b1)
    _Float16* pyh  = pxbh + (size_t)NS * HID;             // 1024*256 f16 (pre_y)
    _Float16* W1t  = pyh  + (size_t)NS * HID;             // 256*256 f16 (W1 transposed)
    _Float16* xh   = W1t  + (size_t)HID * HID;            // 1024*128 f16
    _Float16* yh   = xh   + (size_t)NS * XDIM;            // 1024*128 f16

    k0_w1t<<<dim3(HID), 256, 0, stream>>>(W1, W1t);
    k0_cvt<<<dim3(128), 256, 0, stream>>>(x, y, xh, yh);
    k1_gemm<<<dim3(NS / 16, HID / 16, 2), 32, 0, stream>>>(xh, yh, W1t, b1, pxbh, pyh);
    k2_pair<<<dim3(NS / 8), 256, 0, stream>>>(pxbh, pyh, W2, b2, term);
    k3_reduce<<<dim3(1), 256, 0, stream>>>(term, (float*)d_out);
}
